// Self_Attention_New1_51977694216345
// MI455X (gfx1250) — matrix operand order repaired and checked by reading, not compiled, not run on hardware
//
#include <hip/hip_runtime.h>
#include <hip/hip_bf16.h>

// ---------------------------------------------------------------------------
// Problem constants (from reference): B=32, S=512, H=768, L=512 (== S), C=3
// ---------------------------------------------------------------------------
#define Bn   32
#define Sn   512
#define Hn   768
#define Cn   3
#define BSn  (Bn * Sn)          // 16384 token rows
#define SHn  ((long)Sn * Hn)    // per-batch activation stride
#define SSn  ((long)Sn * Sn)    // per-batch token-attention stride
#define HHn  ((long)Hn * Hn)    // per-batch feature-attention stride

typedef __attribute__((ext_vector_type(16))) __bf16 v16bf;
typedef __attribute__((ext_vector_type(8)))  __bf16 v8bf;
typedef __attribute__((ext_vector_type(8)))  float  v8f;

// ---------------------------------------------------------------------------
// CDNA5 async global -> LDS copy (16 bytes per lane), ASYNCcnt-tracked.
// LDS destination address = low 32 bits of the flat shared-aperture address.
// ---------------------------------------------------------------------------
__device__ __forceinline__ unsigned lds_off_of(const void* p) {
    return (unsigned)(unsigned long long)p;
}
__device__ __forceinline__ void async_g2l_b128(unsigned lds_off, const void* g) {
    asm volatile("global_load_async_to_lds_b128 %0, %1, off"
                 :: "v"(lds_off), "v"((unsigned long long)g) : "memory");
}
__device__ __forceinline__ void wait_async0() {
    asm volatile("s_wait_asynccnt 0" ::: "memory");
}

// ---------------------------------------------------------------------------
// fp32 -> bf16 conversion
// ---------------------------------------------------------------------------
__global__ void f2bf_kernel(const float* __restrict__ x, __bf16* __restrict__ y, long n) {
    long i = (long)blockIdx.x * blockDim.x + threadIdx.x;
    if (i < n) y[i] = (__bf16)x[i];
}

// fp32 [H,H] -> bf16 transposed [H,H]; LDS-tiled so both sides stay coalesced
__global__ __launch_bounds__(256)
void f2bf_t_kernel(const float* __restrict__ x, __bf16* __restrict__ y) {
    __shared__ __bf16 t[32][33];
    const int bx = blockIdx.x, by = blockIdx.y;
    const int tx = threadIdx.x & 31, ty = threadIdx.x >> 5;   // 32 x 8
    #pragma unroll
    for (int r = 0; r < 32; r += 8)
        t[ty + r][tx] = (__bf16)x[(long)(by * 32 + ty + r) * Hn + bx * 32 + tx];
    __syncthreads();
    #pragma unroll
    for (int r = 0; r < 32; r += 8)
        y[(long)(bx * 32 + ty + r) * Hn + by * 32 + tx] = t[tx][ty + r];
}

// ---------------------------------------------------------------------------
// Tiled WMMA GEMM, bf16 inputs, fp32 accumulate.
//   C[m,n] = alpha * sum_k A(m,k) * B(k,n) + bias[n]
// LAYA: 0 -> A memory is [M,K] row-major (direct, async-fed)
//       1 -> A memory is [K,M] row-major (transpose on LDS fill)
// LAYB: 0 -> B memory is [K,N] row-major (transpose on LDS fill)
//       1 -> B memory is [N,K] row-major (direct, async-fed)
// Block tile 64x64x32, 128 threads = 4 waves; each wave owns a 16-row strip
// and 4 accumulators (16x16 each) spanning the 64-wide N tile.
// ---------------------------------------------------------------------------
template <int LAYA, int LAYB, bool OUTBF>
__global__ __launch_bounds__(128)
void gemm_wmma_bf16(const __bf16* __restrict__ A, const __bf16* __restrict__ B,
                    const float* __restrict__ bias, void* __restrict__ Cout,
                    int Kd, int lda, int ldb, int ldc,
                    long sA, long sB, long sC, float alpha)
{
    constexpr int BM = 64, BN = 64, BK = 32;
    __shared__ __align__(16) __bf16 lA[BM][BK];   // row m contiguous in k
    __shared__ __align__(16) __bf16 lB[BN][BK];   // row n contiguous in k

    const int tid  = threadIdx.x;
    const int wave = tid >> 5;
    const int lane = tid & 31;
    const int bz   = blockIdx.z;
    const int m0g  = blockIdx.y * BM;
    const int n0g  = blockIdx.x * BN;
    const __bf16* Ab = A + (long)bz * sA;
    const __bf16* Bb = B + (long)bz * sB;

    v8f acc[4] = {};

    for (int kk = 0; kk < Kd; kk += BK) {
        // ---- stage A tile into LDS ----
        if constexpr (LAYA == 0) {
            #pragma unroll
            for (int i = 0; i < 2; ++i) {                  // 256 x 16B chunks
                int v = tid + i * 128;
                int row = v >> 2, c8 = (v & 3) << 3;
                async_g2l_b128(lds_off_of(&lA[row][c8]),
                               Ab + (long)(m0g + row) * lda + kk + c8);
            }
        } else {
            #pragma unroll
            for (int i = 0; i < 2; ++i) {                  // 8-wide vector rows,
                int c = tid + i * 128;                     // scatter-transpose
                int k = c >> 3, n0 = (c & 7) << 3;
                v8bf r = *reinterpret_cast<const v8bf*>(
                    Ab + (long)(kk + k) * lda + m0g + n0);
                #pragma unroll
                for (int j = 0; j < 8; ++j) lA[n0 + j][k] = r[j];
            }
        }
        // ---- stage B tile into LDS ----
        if constexpr (LAYB == 1) {
            #pragma unroll
            for (int i = 0; i < 2; ++i) {
                int v = tid + i * 128;
                int row = v >> 2, c8 = (v & 3) << 3;
                async_g2l_b128(lds_off_of(&lB[row][c8]),
                               Bb + (long)(n0g + row) * ldb + kk + c8);
            }
        } else {
            #pragma unroll
            for (int i = 0; i < 2; ++i) {
                int c = tid + i * 128;
                int k = c >> 3, n0 = (c & 7) << 3;
                v8bf r = *reinterpret_cast<const v8bf*>(
                    Bb + (long)(kk + k) * ldb + n0g + n0);
                #pragma unroll
                for (int j = 0; j < 8; ++j) lB[n0 + j][k] = r[j];
            }
        }
        if constexpr (LAYA == 0 || LAYB == 1) wait_async0();
        __syncthreads();

        // ---- A fragment (ISA 16-bit A 16x32 layout) ----
        // lanes 0-15: row M=lane,    elems = K{0..7} ++ K{16..23}
        // lanes16-31: row M=lane-16, elems = K{8..15} ++ K{24..31}
        const int mr  = (wave << 4) + (lane & 15);
        const int kbA = (lane < 16) ? 0 : 8;
        v8bf alo = *reinterpret_cast<const v8bf*>(&lA[mr][kbA]);
        v8bf ahi = *reinterpret_cast<const v8bf*>(&lA[mr][kbA + 16]);
        v16bf afrag = __builtin_shufflevector(alo, ahi,
            0, 1, 2, 3, 4, 5, 6, 7, 8, 9, 10, 11, 12, 13, 14, 15);

        #pragma unroll
        for (int nt = 0; nt < 4; ++nt) {
            // ---- B fragment: column n, same K order as A fragment ----
            const int nr  = (nt << 4) + (lane & 15);
            const int kbB = (lane < 16) ? 0 : 8;
            v8bf blo = *reinterpret_cast<const v8bf*>(&lB[nr][kbB]);
            v8bf bhi = *reinterpret_cast<const v8bf*>(&lB[nr][kbB + 16]);
            v16bf bfrag = __builtin_shufflevector(blo, bhi,
                0, 1, 2, 3, 4, 5, 6, 7, 8, 9, 10, 11, 12, 13, 14, 15);

            acc[nt] = __builtin_amdgcn_wmma_f32_16x16x32_bf16(
                false, afrag, false, bfrag, (short)0, acc[nt], false, false);
        }
        __syncthreads();
    }

    // ---- epilogue: C/D layout VGPR r -> M=r (lanes 0-15) / M=r+8 (16-31) ----
    const int rbase = m0g + (wave << 4) + ((lane & 16) ? 8 : 0);
    #pragma unroll
    for (int nt = 0; nt < 4; ++nt) {
        const int c  = n0g + (nt << 4) + (lane & 15);
        const float bv = bias ? bias[c] : 0.0f;
        #pragma unroll
        for (int r = 0; r < 8; ++r) {
            const int  rowg = rbase + r;
            const float val = alpha * acc[nt][r] + bv;
            const long idx  = (long)bz * sC + (long)rowg * ldc + c;
            if constexpr (OUTBF) ((__bf16*)Cout)[idx] = (__bf16)val;
            else                 ((float*)Cout)[idx]  = val;
        }
    }
}

// ---------------------------------------------------------------------------
// Row softmax (ncols in {512, 768}), fp32 in -> bf16 probabilities out
// ---------------------------------------------------------------------------
__global__ __launch_bounds__(256)
void softmax_rows_kernel(const float* __restrict__ X, __bf16* __restrict__ P, int ncols)
{
    __shared__ float red[256];
    const int  t   = threadIdx.x;
    const long row = blockIdx.x;
    const float* x = X + row * (long)ncols;
    const int iters = ncols >> 8;                 // 2 or 3
    float vals[3];
    float mx = -3.4e38f;
    for (int i = 0; i < iters; ++i) { vals[i] = x[t + (i << 8)]; mx = fmaxf(mx, vals[i]); }
    red[t] = mx; __syncthreads();
    for (int o = 128; o > 0; o >>= 1) { if (t < o) red[t] = fmaxf(red[t], red[t + o]); __syncthreads(); }
    mx = red[0]; __syncthreads();
    float s = 0.0f;
    for (int i = 0; i < iters; ++i) { vals[i] = __expf(vals[i] - mx); s += vals[i]; }
    red[t] = s; __syncthreads();
    for (int o = 128; o > 0; o >>= 1) { if (t < o) red[t] += red[t + o]; __syncthreads(); }
    const float inv = 1.0f / red[0];
    __bf16* p = P + row * (long)ncols;
    for (int i = 0; i < iters; ++i) p[t + (i << 8)] = (__bf16)(vals[i] * inv);
}

// ---------------------------------------------------------------------------
// Branch statistics: per batch, column means over S plus scalar mean/var over
// S*H (that's all LayerNorm+mean-pool needs, since LN affine is identity).
// MODE 0: X = BR buffer          MODE 1: X = tokens (no LN branch)
// MODE 2: X = g[b,s]  * tokens   MODE 3: X = f[b,h] * tokens
// ---------------------------------------------------------------------------
template <int MODE>
__global__ __launch_bounds__(256)
void stats_kernel(const float* __restrict__ X, const float* __restrict__ tok,
                  const float* __restrict__ gate, float* __restrict__ colmean,
                  float* __restrict__ mout, float* __restrict__ vout)
{
    const int b = blockIdx.x, t = threadIdx.x;
    __shared__ float red[256];
    __shared__ float red2[256];
    float acc[3] = {0.f, 0.f, 0.f};
    float s = 0.f, sq = 0.f;
    float fg[3] = {1.f, 1.f, 1.f};
    if (MODE == 3) {
        #pragma unroll
        for (int j = 0; j < 3; ++j) fg[j] = gate[b * Hn + t + (j << 8)];
    }
    const float* base = ((MODE == 0) ? X : tok) + (long)b * SHn;
    for (int si = 0; si < Sn; ++si) {
        const float* rowp = base + (long)si * Hn;
        const float gs = (MODE == 2) ? gate[b * Sn + si] : 1.0f;
        #pragma unroll
        for (int j = 0; j < 3; ++j) {
            float v = rowp[t + (j << 8)];
            if (MODE == 2) v *= gs;
            if (MODE == 3) v *= fg[j];
            acc[j] += v; s += v; sq += v * v;
        }
    }
    #pragma unroll
    for (int j = 0; j < 3; ++j)
        colmean[(long)b * Hn + t + (j << 8)] = acc[j] * (1.0f / Sn);
    red[t] = s; red2[t] = sq; __syncthreads();
    for (int o = 128; o > 0; o >>= 1) {
        if (t < o) { red[t] += red[t + o]; red2[t] += red2[t + o]; }
        __syncthreads();
    }
    if (t == 0) {
        const float m = red[0] / (float)(SHn);
        mout[b] = m;
        vout[b] = red2[0] / (float)(SHn) - m * m;
    }
}

// BR = sigmoid(BR) * tokens   (TGSA gate)
__global__ void sigmul_kernel(float* __restrict__ br, const float* __restrict__ tok, long n)
{
    long i = (long)blockIdx.x * blockDim.x + threadIdx.x;
    if (i < n) br[i] = tok[i] / (1.0f + __expf(-br[i]));
}

// g[b,s] = relu(tokens[b,s,:] . w_ts + b_ts), one wave per row
__global__ __launch_bounds__(128)
void gate_ts_kernel(const float* __restrict__ tok, const float* __restrict__ w,
                    const float* __restrict__ bptr, float* __restrict__ g)
{
    const int row  = blockIdx.x * 4 + (threadIdx.x >> 5);
    const int lane = threadIdx.x & 31;
    const float* x = tok + (long)row * Hn;
    float s = 0.f;
    for (int h = lane; h < Hn; h += 32) s += x[h] * w[h];
    for (int o = 16; o > 0; o >>= 1) s += __shfl_xor(s, o, 32);
    if (lane == 0) g[row] = fmaxf(s + bptr[0], 0.0f);
}

// f[b,h] = relu(sum_s tokens[b,s,h] * w_fs[s] + b_fs)  (L == S, pad is no-op)
__global__ __launch_bounds__(256)
void gate_fs_kernel(const float* __restrict__ tok, const float* __restrict__ w,
                    const float* __restrict__ bptr, float* __restrict__ f)
{
    const int b = blockIdx.x, t = threadIdx.x;
    float acc[3] = {0.f, 0.f, 0.f};
    const float* base = tok + (long)b * SHn;
    for (int s = 0; s < Sn; ++s) {
        const float ws = w[s];
        #pragma unroll
        for (int j = 0; j < 3; ++j)
            acc[j] += base[(long)s * Hn + t + (j << 8)] * ws;
    }
    #pragma unroll
    for (int j = 0; j < 3; ++j)
        f[b * Hn + t + (j << 8)] = fmaxf(acc[j] + bptr[0], 0.0f);
}

// pooled[b, j*H + h]: branch order tokens,TSA,FSA,TGSA,FGSA(=FSG),TSGSA,FSGSA
__global__ __launch_bounds__(256)
void pooled_kernel(const float* __restrict__ colmean, const float* __restrict__ m_all,
                   const float* __restrict__ v_all, float* __restrict__ pooled)
{
    const int b   = blockIdx.y;
    const int idx = blockIdx.x * 256 + threadIdx.x;   // 0..5375
    const int j   = idx / Hn, h = idx % Hn;
    const int slot_map[7] = {0, 1, 2, 3, 4, 5, 4};    // FGSA aliases FSGSA stats
    const int slot = slot_map[j];
    const float cm = colmean[(long)slot * Bn * Hn + (long)b * Hn + h];
    float outv;
    if (j == 0) outv = cm;
    else {
        const float m = m_all[slot * Bn + b];
        const float v = v_all[slot * Bn + b];
        outv = (cm - m) * rsqrtf(v + 1e-8f);
    }
    pooled[(long)b * (7 * Hn) + idx] = outv;
}

__global__ __launch_bounds__(256)
void fnn1_kernel(const float* __restrict__ pooled, const float* __restrict__ W1,
                 const float* __restrict__ b1, float* __restrict__ hidden)
{
    const int b = blockIdx.y;
    const int n = blockIdx.x * 256 + threadIdx.x;
    const float* p = pooled + (long)b * (7 * Hn);
    float s = b1[n];
    for (int i = 0; i < 7 * Hn; ++i) s += p[i] * W1[(long)i * Hn + n];
    hidden[(long)b * Hn + n] = s;
}

__global__ void fnn2_kernel(const float* __restrict__ hidden, const float* __restrict__ W2,
                            const float* __restrict__ b2, float* __restrict__ out)
{
    const int t = threadIdx.x;
    if (t >= Bn * Cn) return;
    const int b = t / Cn, c = t % Cn;
    float s = b2[c];
    const float* h = hidden + (long)b * Hn;
    for (int n = 0; n < Hn; ++n) s += h[n] * W2[n * Cn + c];
    out[t] = s;
}

// ---------------------------------------------------------------------------
// Host driver
// ---------------------------------------------------------------------------
extern "C" void kernel_launch(void* const* d_in, const int* in_sizes, int n_in,
                              void* d_out, int out_size, void* d_ws, size_t ws_size,
                              hipStream_t stream)
{
    const float* tokens = (const float*)d_in[0];
    const float* Wq  = (const float*)d_in[1];  const float* bq  = (const float*)d_in[2];
    const float* Wk  = (const float*)d_in[3];  const float* bk  = (const float*)d_in[4];
    const float* Wv  = (const float*)d_in[5];  const float* bv  = (const float*)d_in[6];
    const float* Wts = (const float*)d_in[7];  const float* bts = (const float*)d_in[8];
    const float* Wfs = (const float*)d_in[9];  const float* bfs = (const float*)d_in[10];
    const float* Wtg = (const float*)d_in[11]; const float* btg = (const float*)d_in[12];
    const float* W1  = (const float*)d_in[13]; const float* b1  = (const float*)d_in[14];
    const float* W2  = (const float*)d_in[15]; const float* b2  = (const float*)d_in[16];
    float* out = (float*)d_out;

    // ---- workspace carve-up (bytes) ----
    char* ws = (char*)d_ws;
    size_t off = 0;
    auto alloc = [&](size_t bytes) { char* p = ws + off; off = (off + bytes + 255) & ~(size_t)255; return p; };
    __bf16* tokbf = (__bf16*)alloc((size_t)BSn * Hn * 2);
    __bf16* wqT   = (__bf16*)alloc((size_t)HHn * 2);         // W^T, bf16
    __bf16* wkT   = (__bf16*)alloc((size_t)HHn * 2);
    __bf16* wvT   = (__bf16*)alloc((size_t)HHn * 2);
    __bf16* wtgT  = (__bf16*)alloc((size_t)HHn * 2);
    __bf16* qbf   = (__bf16*)alloc((size_t)BSn * Hn * 2);
    __bf16* kbf   = (__bf16*)alloc((size_t)BSn * Hn * 2);
    __bf16* vbf   = (__bf16*)alloc((size_t)BSn * Hn * 2);
    float*  sc    = (float*) alloc((size_t)Bn * HHn * 4);    // scores (reused)
    __bf16* pbf   = (__bf16*)alloc((size_t)Bn * HHn * 2);    // probabilities (reused)
    float*  br    = (float*) alloc((size_t)BSn * Hn * 4);    // branch buffer (reused)
    float*  colmean = (float*)alloc((size_t)6 * Bn * Hn * 4);
    float*  m_all   = (float*)alloc((size_t)6 * Bn * 4);
    float*  v_all   = (float*)alloc((size_t)6 * Bn * 4);
    float*  g_ts    = (float*)alloc((size_t)BSn * 4);
    float*  f_fs    = (float*)alloc((size_t)Bn * Hn * 4);
    float*  pooled  = (float*)alloc((size_t)Bn * 7 * Hn * 4);
    float*  hidden  = (float*)alloc((size_t)Bn * Hn * 4);

    const float scale = 1.0f / sqrtf((float)Hn);

    // ---- fp32 -> bf16 (weights transposed so GEMM B-operand is async-direct) ----
    {
        long n = (long)BSn * Hn;
        f2bf_kernel<<<(n + 255) / 256, 256, 0, stream>>>(tokens, tokbf, n);
        dim3 gt(Hn / 32, Hn / 32);
        f2bf_t_kernel<<<gt, 256, 0, stream>>>(Wq, wqT);
        f2bf_t_kernel<<<gt, 256, 0, stream>>>(Wk, wkT);
        f2bf_t_kernel<<<gt, 256, 0, stream>>>(Wv, wvT);
        f2bf_t_kernel<<<gt, 256, 0, stream>>>(Wtg, wtgT);
    }

    // ---- Q/K/V projections: [16384,768] @ [768,768] + bias -> bf16 ----
    {
        dim3 g(Hn / 64, BSn / 64, 1);
        gemm_wmma_bf16<0, 1, true><<<g, 128, 0, stream>>>(tokbf, wqT, bq, qbf,
            Hn, Hn, Hn, Hn, 0, 0, 0, 1.0f);
        gemm_wmma_bf16<0, 1, true><<<g, 128, 0, stream>>>(tokbf, wkT, bk, kbf,
            Hn, Hn, Hn, Hn, 0, 0, 0, 1.0f);
        gemm_wmma_bf16<0, 1, true><<<g, 128, 0, stream>>>(tokbf, wvT, bv, vbf,
            Hn, Hn, Hn, Hn, 0, 0, 0, 1.0f);
    }

    // ---- token attention: sc = scale * Q @ K^T (per batch), softmax, @ V ----
    {
        dim3 g(Sn / 64, Sn / 64, Bn);
        gemm_wmma_bf16<0, 1, false><<<g, 128, 0, stream>>>(qbf, kbf, nullptr, sc,
            Hn, Hn, Hn, Sn, SHn, SHn, SSn, scale);
        softmax_rows_kernel<<<Bn * Sn, 256, 0, stream>>>(sc, pbf, Sn);
        dim3 g2(Hn / 64, Sn / 64, Bn);
        gemm_wmma_bf16<0, 0, false><<<g2, 128, 0, stream>>>(pbf, vbf, nullptr, br,
            Sn, Sn, Hn, Hn, SSn, SHn, SHn, 1.0f);
        stats_kernel<0><<<Bn, 256, 0, stream>>>(br, tokens, nullptr,
            colmean + 1L * Bn * Hn, m_all + 1 * Bn, v_all + 1 * Bn);       // TSA
    }

    // ---- feature attention: sc = scale * Q^T @ K (per batch), softmax, V @ ----
    {
        dim3 g(Hn / 64, Hn / 64, Bn);
        gemm_wmma_bf16<1, 0, false><<<g, 128, 0, stream>>>(qbf, kbf, nullptr, sc,
            Sn, Hn, Hn, Hn, SHn, SHn, HHn, scale);
        softmax_rows_kernel<<<Bn * Hn, 256, 0, stream>>>(sc, pbf, Hn);
        dim3 g2(Hn / 64, Sn / 64, Bn);
        gemm_wmma_bf16<0, 0, false><<<g2, 128, 0, stream>>>(vbf, pbf, nullptr, br,
            Hn, Hn, Hn, Hn, SHn, HHn, SHn, 1.0f);
        stats_kernel<0><<<Bn, 256, 0, stream>>>(br, tokens, nullptr,
            colmean + 2L * Bn * Hn, m_all + 2 * Bn, v_all + 2 * Bn);       // FSA
    }

    // ---- TGSA: sigmoid(tokens @ Wtg + btg) * tokens ----
    {
        dim3 g(Hn / 64, BSn / 64, 1);
        gemm_wmma_bf16<0, 1, false><<<g, 128, 0, stream>>>(tokbf, wtgT, btg, br,
            Hn, Hn, Hn, Hn, 0, 0, 0, 1.0f);
        long n = (long)BSn * Hn;
        sigmul_kernel<<<(n + 255) / 256, 256, 0, stream>>>(br, tokens, n);
        stats_kernel<0><<<Bn, 256, 0, stream>>>(br, tokens, nullptr,
            colmean + 3L * Bn * Hn, m_all + 3 * Bn, v_all + 3 * Bn);       // TGSA
    }

    // ---- scalar gates + gated statistics (TSGSA / FSGSA), token means ----
    gate_ts_kernel<<<BSn / 4, 128, 0, stream>>>(tokens, Wts, bts, g_ts);
    gate_fs_kernel<<<Bn, 256, 0, stream>>>(tokens, Wfs, bfs, f_fs);
    stats_kernel<3><<<Bn, 256, 0, stream>>>(nullptr, tokens, f_fs,
        colmean + 4L * Bn * Hn, m_all + 4 * Bn, v_all + 4 * Bn);           // FSGSA
    stats_kernel<2><<<Bn, 256, 0, stream>>>(nullptr, tokens, g_ts,
        colmean + 5L * Bn * Hn, m_all + 5 * Bn, v_all + 5 * Bn);           // TSGSA
    stats_kernel<1><<<Bn, 256, 0, stream>>>(nullptr, tokens, nullptr,
        colmean + 0L * Bn * Hn, m_all + 0 * Bn, v_all + 0 * Bn);           // tokens

    // ---- pool + FNN head ----
    {
        dim3 g(7 * Hn / 256, Bn);
        pooled_kernel<<<g, 256, 0, stream>>>(colmean, m_all, v_all, pooled);
        dim3 g1(Hn / 256, Bn);
        fnn1_kernel<<<g1, 256, 0, stream>>>(pooled, W1, b1, hidden);
        fnn2_kernel<<<1, 128, 0, stream>>>(hidden, W2, b2, out);
    }
    (void)in_sizes; (void)n_in; (void)out_size; (void)ws_size;
}
